// EfficientAttention_79439715107010
// MI455X (gfx1250) — compile-verified, matrix operand order checked by reading, not run on hardware
//
#include <hip/hip_runtime.h>
#include <stdint.h>

typedef __attribute__((ext_vector_type(16))) __bf16 v16bf;
typedef __attribute__((ext_vector_type(8)))  float  v8f;

__device__ __forceinline__ unsigned short f2bf(float f) {
  union { float f; uint32_t u; } x; x.f = f;
  uint32_t r = x.u + 0x7FFFu + ((x.u >> 16) & 1u);   // round-to-nearest-even
  return (unsigned short)(r >> 16);
}

union Frag16 { uint32_t u[8]; v16bf v; };

// CDNA5 async copy: global -> LDS, 16B per lane, tracked by ASYNCcnt.
__device__ __forceinline__ void async_copy_b128(uint32_t lds_addr, const void* gptr) {
  asm volatile("global_load_async_to_lds_b128 %0, %1, off"
               :: "v"(lds_addr), "v"((uint64_t)(uintptr_t)gptr)
               : "memory");
}
__device__ __forceinline__ void wait_asynccnt0() {
  asm volatile("s_wait_asynccnt 0x0" ::: "memory");
}

#define BM 128
#define BN 128
#define BK 32
#define LDSS 40   // 32 + 8 pad, in bf16 elements (row stride 80B, 16B-aligned)

// C[M,N] = alpha * A[M,K] x B^T + bias ; B stored [N,K] if B_NK else [K,N].
// Batched via z = b*H + h with independent b/h strides for A/B/C.
template<bool A_F32, bool B_NK, bool OUT_BF16>
__global__ void __launch_bounds__(256)
gemm_wmma(const void* __restrict__ Ap, const unsigned short* __restrict__ Bp,
          const float* __restrict__ bias, void* __restrict__ Cp,
          int M, int N, int K, int lda, int ldb, int ldc,
          float alpha, int H,
          long long sAb, long long sAh, long long sBb, long long sBh,
          long long sCb, long long sCh)
{
  __shared__ unsigned short Alds[BM][LDSS];
  __shared__ unsigned short Blds[BN][LDSS];

  const int tid  = threadIdx.x;
  const int wave = tid >> 5;
  const int lane = tid & 31;
  const int hlf  = lane >> 4;      // which half-wave
  const int r16  = lane & 15;
  const int wm   = wave & 3;       // 4 waves along M (32 rows each)
  const int wn   = wave >> 2;      // 2 waves along N (64 cols each)

  const int z  = blockIdx.z;
  const int bI = z / H, hI = z % H;
  const long long offA = (long long)bI * sAb + (long long)hI * sAh;
  const long long offB = (long long)bI * sBb + (long long)hI * sBh;
  const long long offC = (long long)bI * sCb + (long long)hI * sCh;

  const int m0 = blockIdx.y * BM;
  const int n0 = blockIdx.x * BN;

  v8f acc[2][4];
#pragma unroll
  for (int i = 0; i < 2; ++i)
#pragma unroll
    for (int j = 0; j < 4; ++j) acc[i][j] = (v8f){0,0,0,0,0,0,0,0};

  const int kbase = hlf * 8;   // ISA 16-bit A layout: lo lanes K0-7/16-23, hi lanes K8-15/24-31

  // Per-thread staging coordinates (contiguous 32B per thread for the async path)
  const int srow = tid >> 1;            // 0..127
  const int skk  = (tid & 1) * 16;      // 0 or 16

  for (int k0 = 0; k0 < K; k0 += BK) {
    // ---- stage A tile [BM x BK] ----
    if (A_F32) {
      const float* A = (const float*)Ap + offA + (long long)(m0 + srow) * lda + k0 + skk;
#pragma unroll
      for (int i = 0; i < 16; ++i) Alds[srow][skk + i] = f2bf(A[i]);
    } else {
      const unsigned short* A =
          (const unsigned short*)Ap + offA + (long long)(m0 + srow) * lda + k0 + skk;
      const uint32_t la = (uint32_t)(uintptr_t)&Alds[srow][skk];
      async_copy_b128(la,      A);
      async_copy_b128(la + 16, A + 8);
    }
    // ---- stage B tile as Blds[n][k] ----
    if (B_NK) {
      const unsigned short* B = Bp + offB + (long long)(n0 + srow) * ldb + k0 + skk;
      const uint32_t lb = (uint32_t)(uintptr_t)&Blds[srow][skk];
      async_copy_b128(lb,      B);
      async_copy_b128(lb + 16, B + 8);
    } else {
      const int kk   = tid >> 3;         // 0..31
      const int nseg = (tid & 7) * 16;   // 0..112
      const unsigned short* B = Bp + offB + (long long)(k0 + kk) * ldb + n0 + nseg;
#pragma unroll
      for (int i = 0; i < 16; ++i) {
        const int n = nseg + i;
        Blds[n][kk] = (n0 + n < N) ? B[i] : (unsigned short)0;
      }
    }
    if (!A_F32 || B_NK) wait_asynccnt0();
    __syncthreads();

    // ---- build fragments ----
    Frag16 a[2], b[4];
#pragma unroll
    for (int fm = 0; fm < 2; ++fm) {
      const int row = wm * 32 + fm * 16 + r16;
#pragma unroll
      for (int v = 0; v < 4; ++v) {
        a[fm].u[v]     = *(const uint32_t*)&Alds[row][kbase + 2 * v];
        a[fm].u[4 + v] = *(const uint32_t*)&Alds[row][16 + kbase + 2 * v];
      }
    }
#pragma unroll
    for (int fn = 0; fn < 4; ++fn) {
      const int row = wn * 64 + fn * 16 + r16;
#pragma unroll
      for (int v = 0; v < 4; ++v) {
        b[fn].u[v]     = *(const uint32_t*)&Blds[row][kbase + 2 * v];
        b[fn].u[4 + v] = *(const uint32_t*)&Blds[row][16 + kbase + 2 * v];
      }
    }

    // ---- 8 WMMAs per k-step ----
#pragma unroll
    for (int fm = 0; fm < 2; ++fm)
#pragma unroll
      for (int fn = 0; fn < 4; ++fn)
        acc[fm][fn] = __builtin_amdgcn_wmma_f32_16x16x32_bf16(
            false, a[fm].v, false, b[fn].v, (short)0, acc[fm][fn], false, false);
    __syncthreads();
  }

  // ---- epilogue: C layout = lane N=lane%16, VGPR rr -> M=rr+8*(lane>=16) ----
#pragma unroll
  for (int fm = 0; fm < 2; ++fm) {
#pragma unroll
    for (int fn = 0; fn < 4; ++fn) {
      const int gn = n0 + wn * 64 + fn * 16 + r16;
      if (gn >= N) continue;
      const float bv = bias ? bias[gn] : 0.0f;
#pragma unroll
      for (int rr = 0; rr < 8; ++rr) {
        const int gm = m0 + wm * 32 + fm * 16 + rr + 8 * hlf;
        const float val = acc[fm][fn][rr] * alpha + bv;
        if (OUT_BF16)
          ((unsigned short*)Cp)[offC + (long long)gm * ldc + gn] = f2bf(val);
        else
          ((float*)Cp)[offC + (long long)gm * ldc + gn] = val;
      }
    }
  }
}

__global__ void __launch_bounds__(256)
softmax_rows(float* __restrict__ attn, int S)
{
  const long long row = blockIdx.x;
  float* p = attn + row * S;
  const int tid = threadIdx.x;
  __shared__ float red[256];

  float m = -3.4e38f;
  for (int i = tid; i < S; i += 256) m = fmaxf(m, p[i]);
  red[tid] = m; __syncthreads();
  for (int s = 128; s > 0; s >>= 1) {
    if (tid < s) red[tid] = fmaxf(red[tid], red[tid + s]);
    __syncthreads();
  }
  m = red[0]; __syncthreads();

  float sum = 0.0f;
  for (int i = tid; i < S; i += 256) { const float e = __expf(p[i] - m); p[i] = e; sum += e; }
  red[tid] = sum; __syncthreads();
  for (int s = 128; s > 0; s >>= 1) {
    if (tid < s) red[tid] += red[tid + s];
    __syncthreads();
  }
  const float inv = 1.0f / red[0];
  for (int i = tid; i < S; i += 256) p[i] *= inv;
}

__global__ void __launch_bounds__(256)
cvt_f32_bf16(const float* __restrict__ in, unsigned short* __restrict__ out, long long n)
{
  long long i = (long long)blockIdx.x * blockDim.x + threadIdx.x;
  const long long stride = (long long)gridDim.x * blockDim.x;
  for (; i < n; i += stride) out[i] = f2bf(in[i]);
}

extern "C" void kernel_launch(void* const* d_in, const int* in_sizes, int n_in,
                              void* d_out, int out_size, void* d_ws, size_t ws_size,
                              hipStream_t stream)
{
  (void)in_sizes; (void)n_in; (void)out_size; (void)ws_size;
  const float* q     = (const float*)d_in[0];
  const float* qkv_w = (const float*)d_in[1];
  const float* qkv_b = (const float*)d_in[2];
  const float* out_w = (const float*)d_in[3];
  const float* out_b = (const float*)d_in[4];

  const int Bb = 2, S = 2048, E = 1024, Hh = 16, D = 64;
  const int M  = Bb * S;        // 4096
  const int E3 = 3 * E;         // 3072

  unsigned char* ws = (unsigned char*)d_ws;
  size_t o = 0;
  unsigned short* Xb    = (unsigned short*)(ws + o); o += (size_t)M  * E  * 2;  // 8 MB
  unsigned short* Wqkvb = (unsigned short*)(ws + o); o += (size_t)E3 * E  * 2;  // 6 MB
  unsigned short* Woutb = (unsigned short*)(ws + o); o += (size_t)E  * E  * 2;  // 2 MB
  unsigned short* qkvb  = (unsigned short*)(ws + o); o += (size_t)M  * E3 * 2;  // 24 MB
  unsigned short* ctxb  = (unsigned short*)(ws + o);                            // 8 MB

  float* outp = (float*)d_out;
  float* attn = outp + (size_t)M * E;   // attn region: [B,H,S,S] fp32

  // fp32 -> bf16 staging
  cvt_f32_bf16<<<1024, 256, 0, stream>>>(q,     Xb,    (long long)M  * E);
  cvt_f32_bf16<<<1024, 256, 0, stream>>>(qkv_w, Wqkvb, (long long)E3 * E);
  cvt_f32_bf16<<<1024, 256, 0, stream>>>(out_w, Woutb, (long long)E  * E);

  // 1) fused QKV projection: qkvb[M, 3E] (bf16 out)
  gemm_wmma<false, true, true><<<dim3(E3 / BN, M / BM, 1), 256, 0, stream>>>(
      Xb, Wqkvb, qkv_b, qkvb, M, E3, E, E, E, E3,
      1.0f, 1, 0, 0, 0, 0, 0, 0);

  // 2) scores = (Q Kt)/sqrt(D), fp32, written straight into d_out attn region
  gemm_wmma<false, true, false><<<dim3(S / BN, S / BM, Bb * Hh), 256, 0, stream>>>(
      qkvb, qkvb + E, nullptr, attn, S, S, D, E3, E3, S,
      0.125f, Hh,
      (long long)S * E3, 64LL,
      (long long)S * E3, 64LL,
      (long long)Hh * S * S, (long long)S * S);

  // 3) softmax in place over rows of attn
  softmax_rows<<<Bb * Hh * S, 256, 0, stream>>>(attn, S);

  // 4) ctx = attn @ V  (A fp32 converted while staging; B stored [K,N]); bf16 out
  gemm_wmma<true, false, true><<<dim3(1, S / BM, Bb * Hh), 256, 0, stream>>>(
      attn, qkvb + 2 * E, nullptr, ctxb, S, D, S, S, E3, E,
      1.0f, Hh,
      (long long)Hh * S * S, (long long)S * S,
      (long long)S * E3, 64LL,
      (long long)S * E, 64LL);

  // 5) out = ctx @ Wout^T + b, fp32 into d_out
  gemm_wmma<false, true, false><<<dim3(E / BN, M / BM, 1), 256, 0, stream>>>(
      ctxb, Woutb, out_b, outp, M, E, E, E, E, E,
      1.0f, 1, 0, 0, 0, 0, 0, 0);
}
